// MultiHeadAttention_69475390980253
// MI455X (gfx1250) — matrix operand order repaired and checked by reading, not compiled, not run on hardware
//
#include <hip/hip_runtime.h>
#include <hip/hip_bf16.h>
#include <stdint.h>

// ---------------------------------------------------------------------------
// Types for CDNA5 WMMA (wave32): V_WMMA_F32_16X16X32_BF16
// ---------------------------------------------------------------------------
typedef __bf16 bf16_t;
typedef __attribute__((ext_vector_type(16))) __bf16 v16bf;
typedef __attribute__((ext_vector_type(8)))  __bf16 v8bf;
typedef __attribute__((ext_vector_type(8)))  float  v8f;

union V16U { v16bf v; v8bf h[2]; };

#define HIDDEN 2048
#define SEQ    2048
#define BATCH  2
#define HEADS  16
#define HDIM   128
#define ROWS   (BATCH * SEQ)          // 4096
#define ATTN_SCALE 0.08838834764831845f   // 1/sqrt(128)
#define MASK_VAL  (-1.0e30f)

__device__ __forceinline__ bf16_t f2bf(float f) {
  union { float f; uint32_t u; } in; in.f = f;
  uint32_t u = in.u;
  u += 0x7FFFu + ((u >> 16) & 1u);    // round-to-nearest-even
  union { uint16_t s; bf16_t b; } out; out.s = (uint16_t)(u >> 16);
  return out.b;
}

// ---------------------------------------------------------------------------
// f32 -> bf16 conversion kernel
// ---------------------------------------------------------------------------
__global__ __launch_bounds__(256) void cvt_f32_bf16(const float* __restrict__ src,
                                                    bf16_t* __restrict__ dst, int n) {
  int i = blockIdx.x * 256 + threadIdx.x;
  if (i < n) dst[i] = f2bf(src[i]);
}

// ---------------------------------------------------------------------------
// Batched NT GEMM: per z-slice, C[M,N] = A[M,K] * B[N,K]^T (bf16 in, f32 acc).
// z decomposes as (zo = z / zInner, zi = z % zInner); per-operand offset =
// zo*So + zi*Si, which covers both linear-z batching and the (b,h) head split.
//
// MODE 0: store f32          MODE 1: store bf16
// MODE 2: store bf16 with attention scale + causal mask (m=query, n=key),
//         fully-masked tiles skip the K loop entirely.
// TRIM_K: limit K loop to mBase+128 (P rows are zero past the causal edge).
//
// BM=BN=128, BK=32, 256 threads = 8 waves, wave computes 32x64 (2x4 frags).
// ---------------------------------------------------------------------------
struct GemmArgs {
  const bf16_t* A; const bf16_t* B; void* C;
  int lda, ldb, ldc, K, zInner;
  long long aSo, aSi, bSo, bSi, cSo, cSi;
};

template <int MODE, bool TRIM_K>
__global__ __launch_bounds__(256) void gemm_nt_bf16(GemmArgs g)
{
  __shared__ bf16_t As[128][40];      // 32 cols + 8 pad (16B-aligned rows)
  __shared__ bf16_t Bs[128][40];

  const int z  = blockIdx.z;
  const int zo = z / g.zInner;
  const int zi = z % g.zInner;
  const bf16_t* A = g.A + zo * g.aSo + zi * g.aSi;
  const bf16_t* B = g.B + zo * g.bSo + zi * g.bSi;
  float*  Cf = (float*)g.C  + zo * g.cSo + zi * g.cSi;
  bf16_t* Cb = (bf16_t*)g.C + zo * g.cSo + zi * g.cSi;

  const int tid  = threadIdx.x;
  const int lane = tid & 31;
  const int wid  = tid >> 5;
  const int wm   = wid & 3;           // 4 wave-rows of 32
  const int wn   = wid >> 2;          // 2 wave-cols of 64
  const int mBase = blockIdx.y * 128;
  const int nBase = blockIdx.x * 128;

  const int fr = lane & 15;
  const int fh = lane >> 4;

  // Fully-masked causal tile: skip compute, write mask value.
  if (MODE == 2 && nBase >= mBase + 128) {
    const bf16_t mv = f2bf(MASK_VAL);
#pragma unroll
    for (int i = 0; i < 2; ++i)
#pragma unroll
      for (int j = 0; j < 4; ++j) {
        const int n     = nBase + wn * 64 + j * 16 + fr;
        const int mRow0 = mBase + wm * 32 + i * 16 + fh * 8;
#pragma unroll
        for (int v = 0; v < 8; ++v)
          Cb[(size_t)(mRow0 + v) * g.ldc + n] = mv;
      }
    return;
  }

  const int lr = tid >> 1;            // LDS fill: row 0..127
  const int lc = (tid & 1) << 4;      // segment 0 or 16

  const int Klim = TRIM_K ? (mBase + 128 < g.K ? mBase + 128 : g.K) : g.K;

  v8f acc[2][4] = {};

  for (int k0 = 0; k0 < Klim; k0 += 32) {
    const uint4* ga = (const uint4*)(A + (size_t)(mBase + lr) * g.lda + (k0 + lc));
    uint4 a0 = ga[0], a1 = ga[1];
    const uint4* gb = (const uint4*)(B + (size_t)(nBase + lr) * g.ldb + (k0 + lc));
    uint4 b0 = gb[0], b1 = gb[1];
    __syncthreads();
    *(uint4*)&As[lr][lc]     = a0;
    *(uint4*)&As[lr][lc + 8] = a1;
    *(uint4*)&Bs[lr][lc]     = b0;
    *(uint4*)&Bs[lr][lc + 8] = b1;
    __syncthreads();

    // Fragment layouts per ISA 7.12.2 (wave32, 16-bit):
    // A 16x32: lane m=fr; elems[0..7]=K(fh*8..), elems[8..15]=K(16+fh*8..)
    // B 32x16: lane n=fr; elems[0..7]=K(fh*8..), elems[8..15]=K(16+fh*8..)
    V16U a[2], b[4];
#pragma unroll
    for (int i = 0; i < 2; ++i) {
      const int row = wm * 32 + i * 16 + fr;
      a[i].h[0] = *(const v8bf*)&As[row][fh * 8];
      a[i].h[1] = *(const v8bf*)&As[row][16 + fh * 8];
    }
#pragma unroll
    for (int j = 0; j < 4; ++j) {
      const int row = wn * 64 + j * 16 + fr;
      b[j].h[0] = *(const v8bf*)&Bs[row][fh * 8];
      b[j].h[1] = *(const v8bf*)&Bs[row][16 + fh * 8];
    }
#pragma unroll
    for (int i = 0; i < 2; ++i)
#pragma unroll
      for (int j = 0; j < 4; ++j)
        acc[i][j] = __builtin_amdgcn_wmma_f32_16x16x32_bf16(
            false, a[i].v, false, b[j].v, (short)0, acc[i][j], false, false);
  }

  // C/D layout: lane -> n = lane%16, vgpr v -> m = v + 8*(lane/16)
#pragma unroll
  for (int i = 0; i < 2; ++i) {
#pragma unroll
    for (int j = 0; j < 4; ++j) {
      const int n     = nBase + wn * 64 + j * 16 + fr;
      const int mRow0 = mBase + wm * 32 + i * 16 + fh * 8;
#pragma unroll
      for (int v = 0; v < 8; ++v) {
        const size_t idx = (size_t)(mRow0 + v) * g.ldc + n;
        if (MODE == 0) {
          Cf[idx] = acc[i][j][v];
        } else if (MODE == 1) {
          Cb[idx] = f2bf(acc[i][j][v]);
        } else {
          float val = acc[i][j][v] * ATTN_SCALE;
          if (n > mRow0 + v) val = MASK_VAL;
          Cb[idx] = f2bf(val);
        }
      }
    }
  }
}

// ---------------------------------------------------------------------------
// In-place softmax over one bf16 row (already scaled + causally masked).
// grid = (SEQ, nSlices); one 256-thread block per row.
// ---------------------------------------------------------------------------
__global__ __launch_bounds__(256) void softmax_inplace(bf16_t* __restrict__ S,
                                                       long long strideZ, int n) {
  bf16_t* row = S + (long long)blockIdx.y * strideZ + (size_t)blockIdx.x * n;
  const int tid = threadIdx.x;
  __shared__ float red[256];

  float mx = -3.4e38f;
  for (int j = tid; j < n; j += 256) mx = fmaxf(mx, (float)row[j]);
  red[tid] = mx;
  __syncthreads();
  for (int s = 128; s > 0; s >>= 1) {
    if (tid < s) red[tid] = fmaxf(red[tid], red[tid + s]);
    __syncthreads();
  }
  mx = red[0];
  __syncthreads();

  float sum = 0.0f;
  for (int j = tid; j < n; j += 256) sum += __expf((float)row[j] - mx);
  red[tid] = sum;
  __syncthreads();
  for (int s = 128; s > 0; s >>= 1) {
    if (tid < s) red[tid] += red[tid + s];
    __syncthreads();
  }
  sum = red[0];

  const float inv = 1.0f / sum;
  for (int j = tid; j < n; j += 256)
    row[j] = f2bf(__expf((float)row[j] - mx) * inv);
}

// ---------------------------------------------------------------------------
// Host side
// ---------------------------------------------------------------------------
extern "C" void kernel_launch(void* const* d_in, const int* in_sizes, int n_in,
                              void* d_out, int out_size, void* d_ws, size_t ws_size,
                              hipStream_t stream) {
  (void)in_sizes; (void)n_in; (void)out_size;
  const float* x  = (const float*)d_in[0];
  const float* Wq = (const float*)d_in[1];
  const float* Wk = (const float*)d_in[2];
  const float* Wv = (const float*)d_in[3];
  const float* Wo = (const float*)d_in[4];
  float* out = (float*)d_out;

  char* ws = (char*)d_ws;
  size_t off = 0;
  auto alloc = [&](size_t bytes) -> void* {
    void* p = ws + off;
    off += (bytes + 255) & ~(size_t)255;
    return p;
  };
  const size_t act = (size_t)ROWS * HIDDEN;     // 8M elements
  const size_t wel = (size_t)HIDDEN * HIDDEN;   // 4M elements
  const long long SS = (long long)SEQ * SEQ;    // score slice elements
  const long long SH = (long long)SEQ * HIDDEN;
  const long long DS = (long long)HDIM * SEQ;

  bf16_t* xb  = (bf16_t*)alloc(act * 2);
  bf16_t* wqb = (bf16_t*)alloc(wel * 2);
  bf16_t* wkb = (bf16_t*)alloc(wel * 2);
  bf16_t* wvb = (bf16_t*)alloc(wel * 2);
  bf16_t* wob = (bf16_t*)alloc(wel * 2);
  bf16_t* Qb  = (bf16_t*)alloc(act * 2);        // [b,s,h,d]
  bf16_t* Kb  = (bf16_t*)alloc(act * 2);        // [b,s,h,d]
  bf16_t* Vt  = (bf16_t*)alloc(act * 2);        // [b,h,d,s]
  bf16_t* Ob  = (bf16_t*)alloc(act * 2);        // [b,s,h,d]

  // Batched score buffer (32 heads) if workspace allows, else 1 reused slice.
  const int nHeads = BATCH * HEADS;             // 32
  const size_t sliceBytes = (size_t)SS * 2;
  const int nSlices = (ws_size >= off + (size_t)nHeads * sliceBytes + 256) ? nHeads : 1;
  bf16_t* Sb = (bf16_t*)alloc((size_t)nSlices * sliceBytes);

  // 1) convert inputs to bf16
  cvt_f32_bf16<<<(int)((act + 255) / 256), 256, 0, stream>>>(x,  xb,  (int)act);
  cvt_f32_bf16<<<(int)((wel + 255) / 256), 256, 0, stream>>>(Wq, wqb, (int)wel);
  cvt_f32_bf16<<<(int)((wel + 255) / 256), 256, 0, stream>>>(Wk, wkb, (int)wel);
  cvt_f32_bf16<<<(int)((wel + 255) / 256), 256, 0, stream>>>(Wv, wvb, (int)wel);
  cvt_f32_bf16<<<(int)((wel + 255) / 256), 256, 0, stream>>>(Wo, wob, (int)wel);

  // 2) Q = x Wq^T, K = x Wk^T  (M=4096, N=2048, K=2048)
  {
    dim3 grid(HIDDEN / 128, ROWS / 128, 1);
    GemmArgs a{xb, wqb, Qb, HIDDEN, HIDDEN, HIDDEN, HIDDEN, 1, 0,0,0,0,0,0};
    gemm_nt_bf16<1, false><<<grid, 256, 0, stream>>>(a);
    a.B = wkb; a.C = Kb;
    gemm_nt_bf16<1, false><<<grid, 256, 0, stream>>>(a);
  }

  // 3) Vt = Wv x^T -> [b,h,d,s]  (per batch z: M=2048(d_tot), N=2048(s), K=2048)
  {
    dim3 grid(SEQ / 128, HIDDEN / 128, BATCH);
    GemmArgs a{wvb, xb, Vt, HIDDEN, HIDDEN, SEQ, HIDDEN, 1,
               0, 0, SH, 0, SH, 0};
    gemm_nt_bf16<1, false><<<grid, 256, 0, stream>>>(a);
  }

  if (nSlices == nHeads) {
    // 4a) batched attention: z = b*16 + h  (zo = b, zi = h)
    // scores: S = Q K^T scaled+masked -> bf16  (M=N=2048, K=128)
    {
      dim3 grid(SEQ / 128, SEQ / 128, nHeads);
      GemmArgs a{Qb, Kb, Sb, HIDDEN, HIDDEN, SEQ, HDIM, HEADS,
                 SH, HDIM, SH, HDIM, (long long)HEADS * SS, SS};
      gemm_nt_bf16<2, false><<<grid, 256, 0, stream>>>(a);
    }
    // softmax in place
    {
      dim3 grid(SEQ, nHeads);
      softmax_inplace<<<grid, 256, 0, stream>>>(Sb, SS, SEQ);
    }
    // O = P Vt^T  (M=2048, N=128, K trimmed causally)
    {
      dim3 grid(HDIM / 128, SEQ / 128, nHeads);
      GemmArgs a{Sb, Vt, Ob, SEQ, SEQ, HIDDEN, SEQ, HEADS,
                 (long long)HEADS * SS, SS,
                 (long long)HEADS * DS, DS,
                 SH, HDIM};
      gemm_nt_bf16<1, true><<<grid, 256, 0, stream>>>(a);
    }
  } else {
    // 4b) sequential fallback (small workspace): one head at a time
    for (int b = 0; b < BATCH; ++b) {
      for (int h = 0; h < HEADS; ++h) {
        const long long qk = (long long)b * SH + (long long)h * HDIM;
        {
          dim3 grid(SEQ / 128, SEQ / 128, 1);
          GemmArgs a{Qb + qk, Kb + qk, Sb, HIDDEN, HIDDEN, SEQ, HDIM, 1,
                     0,0,0,0,0,0};
          gemm_nt_bf16<2, false><<<grid, 256, 0, stream>>>(a);
        }
        softmax_inplace<<<dim3(SEQ, 1), 256, 0, stream>>>(Sb, 0, SEQ);
        {
          dim3 grid(HDIM / 128, SEQ / 128, 1);
          GemmArgs a{Sb, Vt + (long long)(b * HEADS + h) * DS,
                     Ob + (long long)b * SH + (long long)h * HDIM,
                     SEQ, SEQ, HIDDEN, SEQ, 1, 0,0,0,0,0,0};
          gemm_nt_bf16<1, true><<<grid, 256, 0, stream>>>(a);
        }
      }
    }
  }

  // 5) Y = O Wo^T  (M=4096, N=2048, K=2048) -> f32 d_out
  {
    dim3 grid(HIDDEN / 128, ROWS / 128, 1);
    GemmArgs a{Ob, wob, out, HIDDEN, HIDDEN, HIDDEN, HIDDEN, 1, 0,0,0,0,0,0};
    gemm_nt_bf16<0, false><<<grid, 256, 0, stream>>>(a);
  }
}
